// DiplomacyValueNet_49529562857949
// MI455X (gfx1250) — compile-verified, matrix operand order checked by reading, not run on hardware
//
#include <hip/hip_runtime.h>
#include <hip/hip_bf16.h>
#include <math.h>

// ---------------------------------------------------------------------------
// DiplomacyValueNet on MI455X (gfx1250): bf16 WMMA GEMMs + VALU glue kernels.
// ---------------------------------------------------------------------------

typedef __attribute__((ext_vector_type(16))) __bf16 v16bf;
typedef __attribute__((ext_vector_type(8)))  float  v8f;

#define MROWS 165888   // B*N = 2048*81
#define DMODEL 512
#define NNODES 81
#define NHEAD 8
#define DHEAD 64
#define NBATCH 2048
#define FFNDIM 2048
#define FIN 47

// -------------------------------- helpers ----------------------------------

__device__ __forceinline__ float gelu_exact(float x) {
    return 0.5f * x * (1.0f + erff(x * 0.70710678118654752f));
}

// -------------------------- weight transpose+convert ------------------------
// W f32 [K][N] row-major  ->  WT bf16 [N][K] row-major
__global__ void k_transpose_bf16(const float* __restrict__ W,
                                 __bf16* __restrict__ WT, int K, int N) {
    __shared__ float tile[32][33];
    int nb = blockIdx.x * 32, kb = blockIdx.y * 32;
    for (int r = threadIdx.y; r < 32; r += 8) {
        int k = kb + r, n = nb + threadIdx.x;
        tile[r][threadIdx.x] = (k < K && n < N) ? W[(size_t)k * N + n] : 0.0f;
    }
    __syncthreads();
    for (int r = threadIdx.y; r < 32; r += 8) {
        int n = nb + r, k = kb + threadIdx.x;
        if (n < N && k < K) WT[(size_t)n * K + k] = (__bf16)tile[threadIdx.x][r];
    }
}

// flat f32 -> bf16 convert (layout-preserving)
__global__ void k_cvt_bf16(const float* __restrict__ src,
                           __bf16* __restrict__ dst, long total) {
    long i = (long)blockIdx.x * blockDim.x + threadIdx.x;
    if (i < total) dst[i] = (__bf16)src[i];
}

// ------------------------------ WMMA GEMM ----------------------------------
// C[M][N] = act(A[M][K](bf16) @ BT[N][K]^T(bf16) + bias) (+ residual)
// macro tile 128(M) x 256(N), KTILE 32, 8 waves (2x4), 64x64 per wave.
struct FragBF { union { v16bf v; unsigned u[8]; }; };

__global__ __launch_bounds__(256, 1)
void k_gemm_wmma(const __bf16* __restrict__ A, const __bf16* __restrict__ BT,
                 const float* __restrict__ bias, const float* __restrict__ residual,
                 float* __restrict__ outF, __bf16* __restrict__ outB,
                 int M, int N, int K, int act) {
    __shared__ __align__(16) __bf16 As[128][36];   // stride 36 -> 18 dwords/row
    __shared__ __align__(16) __bf16 Bs[256][36];

    const int tid  = threadIdx.x;
    const int lane = tid & 31;
    const int wave = tid >> 5;
    const int wm   = wave >> 2;          // 0..1  (M direction, 64 rows each)
    const int wn   = wave & 3;           // 0..3  (N direction, 64 cols each)
    const int m0   = blockIdx.y * 128;
    const int n0   = blockIdx.x * 256;

    const int mr = lane & 15;            // fragment row / col within 16
    const int kb = (lane >> 4) * 8;      // K base for this half-wave
    const int kdw = kb >> 1;             // dword base within LDS row

    const bool kAligned = ((K & 31) == 0);

    v8f acc[4][4];
    const v8f vzero = 0.0f;
#pragma unroll
    for (int i = 0; i < 4; i++)
#pragma unroll
        for (int j = 0; j < 4; j++) acc[i][j] = vzero;

    for (int ko = 0; ko < K; ko += 32) {
        __syncthreads();
        // ---- stage A tile: 128x32 bf16, thread -> (row, 16-elem half) ----
        {
            int row = tid >> 1;
            int cb  = (tid & 1) * 16;
            int gm  = m0 + row;
            if (kAligned) {
                if (gm < M) {
                    const uint2* s =
                        reinterpret_cast<const uint2*>(A + (size_t)gm * K + ko + cb);
                    uint2* d = reinterpret_cast<uint2*>(&As[row][cb]);
                    d[0] = s[0]; d[1] = s[1]; d[2] = s[2]; d[3] = s[3];
                    // prefetch next K tile of A (global_prefetch_b8 path)
                    if (ko + 32 < K)
                        __builtin_prefetch(A + (size_t)gm * K + ko + 32 + cb, 0, 1);
                } else {
                    uint2 z; z.x = 0u; z.y = 0u;
                    uint2* d = reinterpret_cast<uint2*>(&As[row][cb]);
                    d[0] = z; d[1] = z; d[2] = z; d[3] = z;
                }
            } else {
#pragma unroll
                for (int j = 0; j < 16; j++) {
                    int kk = ko + cb + j;
                    __bf16 val = (__bf16)0.0f;
                    if (gm < M && kk < K) val = A[(size_t)gm * K + kk];
                    As[row][cb + j] = val;
                }
            }
        }
        // ---- stage B tile: 256 rows (N) x 32 (K) from WT[N][K] ----
        {
            int n = tid;                      // 0..255, always < N (N%256==0)
            const __bf16* src = BT + (size_t)(n0 + n) * K + ko;
            if (kAligned) {
                const uint2* s = reinterpret_cast<const uint2*>(src);
                uint2* d = reinterpret_cast<uint2*>(&Bs[n][0]);
#pragma unroll
                for (int j = 0; j < 8; j++) d[j] = s[j];
            } else {
#pragma unroll
                for (int j = 0; j < 32; j++) {
                    int kk = ko + j;
                    Bs[n][j] = (kk < K) ? src[j] : (__bf16)0.0f;
                }
            }
        }
        __syncthreads();

        // ---- load fragments (ISA 16-bit A-matrix layout) and WMMA ----
        FragBF afr[4], bfr[4];
#pragma unroll
        for (int i = 0; i < 4; i++) {
            const unsigned* p =
                reinterpret_cast<const unsigned*>(As[wm * 64 + i * 16 + mr]);
#pragma unroll
            for (int v = 0; v < 8; v++)
                afr[i].u[v] = p[kdw + v + ((v >= 4) ? 4 : 0)];
        }
#pragma unroll
        for (int j = 0; j < 4; j++) {
            const unsigned* p =
                reinterpret_cast<const unsigned*>(Bs[wn * 64 + j * 16 + mr]);
#pragma unroll
            for (int v = 0; v < 8; v++)
                bfr[j].u[v] = p[kdw + v + ((v >= 4) ? 4 : 0)];
        }
#pragma unroll
        for (int i = 0; i < 4; i++)
#pragma unroll
            for (int j = 0; j < 4; j++)
                acc[i][j] = __builtin_amdgcn_wmma_f32_16x16x32_bf16(
                    false, afr[i].v, false, bfr[j].v,
                    (short)0, acc[i][j], false, false);
    }

    // ---- epilogue: bias -> activation -> residual -> store(s) ----
    const int nlane = lane & 15;
    const int mhalf = (lane >> 4) * 8;
#pragma unroll
    for (int i = 0; i < 4; i++) {
        int gmb = m0 + wm * 64 + i * 16 + mhalf;
#pragma unroll
        for (int j = 0; j < 4; j++) {
            int gn = n0 + wn * 64 + j * 16 + nlane;
            float bi = bias ? bias[gn] : 0.0f;
#pragma unroll
            for (int r = 0; r < 8; r++) {
                int gm = gmb + r;
                if (gm < M) {
                    float val = acc[i][j][r] + bi;
                    if (act == 1)      val = gelu_exact(val);
                    else if (act == 2) val = tanhf(val);
                    size_t idx = (size_t)gm * N + gn;
                    if (residual) val += residual[idx];
                    if (outF) outF[idx] = val;
                    if (outB) outB[idx] = (__bf16)val;
                }
            }
        }
    }
}

// ------------------------------- LayerNorm ---------------------------------
// one block (256 thr) per row of 512; optional f32 and/or bf16 output
__global__ void k_layernorm(const float* __restrict__ x,
                            const float* __restrict__ g, const float* __restrict__ b,
                            float* __restrict__ outF, __bf16* __restrict__ outB) {
    __shared__ float red[256];
    int row = blockIdx.x, tid = threadIdx.x;
    const float* xr = x + (size_t)row * DMODEL;
    float v0 = xr[tid], v1 = xr[tid + 256];

    red[tid] = v0 + v1; __syncthreads();
    for (int s = 128; s > 0; s >>= 1) {
        if (tid < s) red[tid] += red[tid + s];
        __syncthreads();
    }
    float mu = red[0] * (1.0f / 512.0f);
    __syncthreads();

    float d0 = v0 - mu, d1 = v1 - mu;
    red[tid] = d0 * d0 + d1 * d1; __syncthreads();
    for (int s = 128; s > 0; s >>= 1) {
        if (tid < s) red[tid] += red[tid + s];
        __syncthreads();
    }
    float inv = rsqrtf(red[0] * (1.0f / 512.0f) + 1e-5f);

    float y0 = d0 * inv * g[tid]       + b[tid];
    float y1 = d1 * inv * g[tid + 256] + b[tid + 256];
    size_t base = (size_t)row * DMODEL;
    if (outF) { outF[base + tid] = y0; outF[base + tid + 256] = y1; }
    if (outB) { outB[base + tid] = (__bf16)y0; outB[base + tid + 256] = (__bf16)y1; }
}

// ------------------------- masked attention (per b,h) -----------------------
__global__ __launch_bounds__(128)
void k_attn(const __bf16* __restrict__ Q, const __bf16* __restrict__ K,
            const __bf16* __restrict__ V, const int* __restrict__ adj,
            __bf16* __restrict__ O) {
    __shared__ __bf16 qs[NNODES * DHEAD];
    __shared__ __bf16 ks[NNODES * DHEAD];
    __shared__ __bf16 vs[NNODES * DHEAD];
    __shared__ unsigned mrow[NNODES][3];
    __shared__ float sc[NNODES][84];

    const int h = blockIdx.x, b = blockIdx.y, tid = threadIdx.x;
    const float scale = 0.125f;  // 1/sqrt(64)

    for (int i = tid; i < NNODES * DHEAD; i += 128) {
        int r = i >> 6, c = i & 63;
        size_t g = ((size_t)(b * NNODES + r)) * DMODEL + h * DHEAD + c;
        qs[i] = Q[g]; ks[i] = K[g]; vs[i] = V[g];
    }
    if (tid < NNODES) {
        unsigned w0 = 0, w1 = 0, w2 = 0;
        for (int c = 0; c < NNODES; c++) {
            int on = (adj[tid * NNODES + c] != 0) || (c == tid);
            if (on) {
                if (c < 32)      w0 |= 1u << c;
                else if (c < 64) w1 |= 1u << (c - 32);
                else             w2 |= 1u << (c - 64);
            }
        }
        mrow[tid][0] = w0; mrow[tid][1] = w1; mrow[tid][2] = w2;
    }
    __syncthreads();

    if (tid < NNODES) {
        const int n = tid;
        float mx = -1e30f;
        for (int m = 0; m < NNODES; m++) {
            float d = 0.0f;
            for (int k = 0; k < DHEAD; k++)
                d += (float)qs[n * DHEAD + k] * (float)ks[m * DHEAD + k];
            bool on = (mrow[n][m >> 5] >> (m & 31)) & 1u;
            float s = on ? d * scale : -1e9f;
            sc[n][m] = s;
            mx = fmaxf(mx, s);
        }
        float ssum = 0.0f;
        for (int m = 0; m < NNODES; m++) {
            float e = __expf(sc[n][m] - mx);
            sc[n][m] = e; ssum += e;
        }
        float inv = 1.0f / ssum;
        for (int d = 0; d < DHEAD; d++) {
            float a = 0.0f;
            for (int m = 0; m < NNODES; m++)
                a += sc[n][m] * (float)vs[m * DHEAD + d];
            O[((size_t)(b * NNODES + n)) * DMODEL + h * DHEAD + d] = (__bf16)(a * inv);
        }
    }
}

// -------------------- context = x + power_embed[power_idx] ------------------
__global__ void k_addpower(const float* __restrict__ X, const float* __restrict__ PE,
                           const int* __restrict__ pidx,
                           float* __restrict__ Cf, __bf16* __restrict__ Cb) {
    long i = (long)blockIdx.x * 256 + threadIdx.x;
    if (i >= (long)MROWS * DMODEL) return;
    int d = (int)(i & 511);
    int m = (int)(i >> 9);
    int b = m / NNODES;
    float v = X[i] + PE[(size_t)pidx[b] * DMODEL + d];
    Cf[i] = v; Cb[i] = (__bf16)v;
}

// ---------------- s[m] = tanh_feat[m] . pw2 + pb2 (dot of 512) --------------
__global__ void k_scoredot(const __bf16* __restrict__ G1, const float* __restrict__ pw2,
                           const float* __restrict__ pb2, float* __restrict__ Sv) {
    __shared__ float red[256];
    int row = blockIdx.x, tid = threadIdx.x;
    const __bf16* g = G1 + (size_t)row * DMODEL;
    red[tid] = (float)g[tid] * pw2[tid] + (float)g[tid + 256] * pw2[tid + 256];
    __syncthreads();
    for (int s = 128; s > 0; s >>= 1) {
        if (tid < s) red[tid] += red[tid + s];
        __syncthreads();
    }
    if (tid == 0) Sv[row] = red[0] + pb2[0];
}

// ------------- softmax over N + weighted pooling: pooled[b][d] --------------
__global__ void k_pool(const float* __restrict__ Sv, const float* __restrict__ Cf,
                       float* __restrict__ pooledF, __bf16* __restrict__ pooledB) {
    __shared__ float w[NNODES];
    int b = blockIdx.x, tid = threadIdx.x;
    if (tid < NNODES) w[tid] = Sv[b * NNODES + tid];
    __syncthreads();
    float mx = -1e30f;
    for (int n = 0; n < NNODES; n++) mx = fmaxf(mx, w[n]);
    __syncthreads();
    if (tid < NNODES) w[tid] = __expf(w[tid] - mx);
    __syncthreads();
    float ssum = 0.0f;
    for (int n = 0; n < NNODES; n++) ssum += w[n];
    float inv = 1.0f / ssum;

    for (int d = tid; d < DMODEL; d += 256) {
        float a = 0.0f;
        for (int n = 0; n < NNODES; n++)
            a += w[n] * Cf[((size_t)(b * NNODES + n)) * DMODEL + d];
        a *= inv;
        pooledF[(size_t)b * DMODEL + d] = a;
        pooledB[(size_t)b * DMODEL + d] = (__bf16)a;
    }
}

// ---------------- out[b][j] = sigmoid(hid[b] . vw2[:,j] + vb2[j]) -----------
__global__ void k_head(const float* __restrict__ hid, const float* __restrict__ vw2,
                       const float* __restrict__ vb2, float* __restrict__ out) {
    int idx = blockIdx.x * 256 + threadIdx.x;  // 8192 total
    if (idx >= NBATCH * 4) return;
    int b = idx >> 2, j = idx & 3;
    float a = vb2[j];
    const float* hr = hid + (size_t)b * 256;
    for (int k = 0; k < 256; k++) a += hr[k] * vw2[k * 4 + j];
    out[idx] = 1.0f / (1.0f + __expf(-a));
}

// ------------------------------ host driver --------------------------------

extern "C" void kernel_launch(void* const* d_in, const int* in_sizes, int n_in,
                              void* d_out, int out_size, void* d_ws, size_t ws_size,
                              hipStream_t stream) {
    const float* board     = (const float*)d_in[0];
    const int*   adj       = (const int*)  d_in[1];
    const int*   pidx      = (const int*)  d_in[2];
    const float* in_w      = (const float*)d_in[3];
    const float* in_b      = (const float*)d_in[4];
    const float* in_ln_g   = (const float*)d_in[5];
    const float* in_ln_b   = (const float*)d_in[6];
    const float* ln1_g     = (const float*)d_in[7];
    const float* ln1_b     = (const float*)d_in[8];
    const float* wq        = (const float*)d_in[9];
    const float* wk        = (const float*)d_in[10];
    const float* wv        = (const float*)d_in[11];
    const float* wo        = (const float*)d_in[12];
    const float* bo        = (const float*)d_in[13];
    const float* ln2_g     = (const float*)d_in[14];
    const float* ln2_b     = (const float*)d_in[15];
    const float* ffn_w1    = (const float*)d_in[16];
    const float* ffn_b1    = (const float*)d_in[17];
    const float* ffn_w2    = (const float*)d_in[18];
    const float* ffn_b2    = (const float*)d_in[19];
    const float* pembed    = (const float*)d_in[20];
    const float* pw1       = (const float*)d_in[21];
    const float* pb1       = (const float*)d_in[22];
    const float* pw2       = (const float*)d_in[23];
    const float* pb2       = (const float*)d_in[24];
    const float* vw1       = (const float*)d_in[25];
    const float* vb1       = (const float*)d_in[26];
    const float* vw2       = (const float*)d_in[27];
    const float* vb2       = (const float*)d_in[28];
    float* out = (float*)d_out;

    // ---- workspace layout ----
    char* ws = (char*)d_ws;
    const size_t szXf = (size_t)MROWS * DMODEL * 4;   // 339,738,624
    const size_t szHb = (size_t)MROWS * DMODEL * 2;   // 169,869,312
    const size_t off_X  = 0;
    const size_t off_H  = szXf;
    const size_t off_S  = szXf + szHb;                // 4*szHb scratch region
    const size_t off_WT = off_S + 4 * szHb;
    const size_t off_SM = off_WT + (size_t)40 * 1024 * 1024;

    float*  X    = (float*)(ws + off_X);
    __bf16* Hbf  = (__bf16*)(ws + off_H);
    // scratch region aliases (phase-disjoint)
    float*  U    = (float*)(ws + off_S);                       // pre-LN input proj
    __bf16* Bbf  = (__bf16*)(ws + off_S + szXf);               // board bf16
    __bf16* Qb   = (__bf16*)(ws + off_S);
    __bf16* Kb   = Qb + (size_t)MROWS * DMODEL;
    __bf16* Vb   = Kb + (size_t)MROWS * DMODEL;
    __bf16* Ob   = Vb + (size_t)MROWS * DMODEL;
    __bf16* Tb   = (__bf16*)(ws + off_S);                      // FFN intermediate
    float*  Cf   = (float*)(ws + off_S);                       // context f32
    __bf16* Cb   = (__bf16*)(ws + off_S + szXf);               // context bf16
    __bf16* G1   = (__bf16*)(ws + off_S + szXf + szHb);        // tanh feat bf16

    // transposed bf16 weights
    __bf16* wt = (__bf16*)(ws + off_WT);
    size_t cur = 0;
    auto walloc = [&](size_t elems) {
        __bf16* p = wt + cur;
        cur += (elems + 127) & ~(size_t)127;
        return p;
    };
    __bf16* inT = walloc((size_t)DMODEL * FIN);
    __bf16 *qT[6], *kT[6], *vT[6], *oT[6], *w1T[6], *w2T[6];
    for (int l = 0; l < 6; l++) {
        qT[l]  = walloc((size_t)DMODEL * DMODEL);
        kT[l]  = walloc((size_t)DMODEL * DMODEL);
        vT[l]  = walloc((size_t)DMODEL * DMODEL);
        oT[l]  = walloc((size_t)DMODEL * DMODEL);
        w1T[l] = walloc((size_t)FFNDIM * DMODEL);
        w2T[l] = walloc((size_t)DMODEL * FFNDIM);
    }
    __bf16* pw1T = walloc((size_t)DMODEL * DMODEL);
    __bf16* vw1T = walloc((size_t)256 * DMODEL);

    // small f32 buffers
    char* sm = ws + off_SM;
    float* Sv      = (float*)sm;                                sm += (size_t)MROWS * 4;
    float* pooledF = (float*)sm;                                sm += (size_t)NBATCH * DMODEL * 4;
    __bf16* pooledB = (__bf16*)sm;                              sm += (size_t)NBATCH * DMODEL * 2;
    float* hid     = (float*)sm;

    auto xpose = [&](const float* W, __bf16* WT, int K, int N) {
        dim3 g((N + 31) / 32, (K + 31) / 32);
        k_transpose_bf16<<<g, dim3(32, 8), 0, stream>>>(W, WT, K, N);
    };
    auto gemm = [&](const __bf16* A, const __bf16* BT, const float* bias,
                    const float* res, float* oF, __bf16* oB,
                    int M, int N, int K, int act) {
        dim3 g(N / 256, (M + 127) / 128);
        k_gemm_wmma<<<g, 256, 0, stream>>>(A, BT, bias, res, oF, oB, M, N, K, act);
    };

    // ---- convert weights (every call: deterministic, graph-capture safe) ----
    xpose(in_w, inT, FIN, DMODEL);
    for (int l = 0; l < 6; l++) {
        xpose(wq + (size_t)l * DMODEL * DMODEL, qT[l], DMODEL, DMODEL);
        xpose(wk + (size_t)l * DMODEL * DMODEL, kT[l], DMODEL, DMODEL);
        xpose(wv + (size_t)l * DMODEL * DMODEL, vT[l], DMODEL, DMODEL);
        xpose(wo + (size_t)l * DMODEL * DMODEL, oT[l], DMODEL, DMODEL);
        xpose(ffn_w1 + (size_t)l * DMODEL * FFNDIM, w1T[l], DMODEL, FFNDIM);
        xpose(ffn_w2 + (size_t)l * FFNDIM * DMODEL, w2T[l], FFNDIM, DMODEL);
    }
    xpose(pw1, pw1T, DMODEL, DMODEL);
    xpose(vw1, vw1T, DMODEL, 256);

    // ---- input projection: X = LN(gelu(board @ in_w + in_b)) ----
    {
        long tot = (long)MROWS * FIN;
        k_cvt_bf16<<<(unsigned)((tot + 255) / 256), 256, 0, stream>>>(board, Bbf, tot);
    }
    gemm(Bbf, inT, in_b, nullptr, U, nullptr, MROWS, DMODEL, FIN, /*gelu*/1);
    k_layernorm<<<MROWS, 256, 0, stream>>>(U, in_ln_g, in_ln_b, X, nullptr);

    // ---- transformer layers ----
    for (int l = 0; l < 6; l++) {
        k_layernorm<<<MROWS, 256, 0, stream>>>(X, ln1_g + l * DMODEL, ln1_b + l * DMODEL,
                                               nullptr, Hbf);
        gemm(Hbf, qT[l], nullptr, nullptr, nullptr, Qb, MROWS, DMODEL, DMODEL, 0);
        gemm(Hbf, kT[l], nullptr, nullptr, nullptr, Kb, MROWS, DMODEL, DMODEL, 0);
        gemm(Hbf, vT[l], nullptr, nullptr, nullptr, Vb, MROWS, DMODEL, DMODEL, 0);
        k_attn<<<dim3(NHEAD, NBATCH), 128, 0, stream>>>(Qb, Kb, Vb, adj, Ob);
        gemm(Ob, oT[l], bo + l * DMODEL, X, X, nullptr, MROWS, DMODEL, DMODEL, 0);
        k_layernorm<<<MROWS, 256, 0, stream>>>(X, ln2_g + l * DMODEL, ln2_b + l * DMODEL,
                                               nullptr, Hbf);
        gemm(Hbf, w1T[l], ffn_b1 + l * FFNDIM, nullptr, nullptr, Tb,
             MROWS, FFNDIM, DMODEL, /*gelu*/1);
        gemm(Tb, w2T[l], ffn_b2 + l * DMODEL, X, X, nullptr,
             MROWS, DMODEL, FFNDIM, 0);
    }

    // ---- power-conditioned attention pooling ----
    k_addpower<<<(unsigned)(((long)MROWS * DMODEL + 255) / 256), 256, 0, stream>>>(
        X, pembed, pidx, Cf, Cb);
    gemm(Cb, pw1T, pb1, nullptr, nullptr, G1, MROWS, DMODEL, DMODEL, /*tanh*/2);
    k_scoredot<<<MROWS, 256, 0, stream>>>(G1, pw2, pb2, Sv);
    k_pool<<<NBATCH, 256, 0, stream>>>(Sv, Cf, pooledF, pooledB);

    // ---- value head ----
    gemm(pooledB, vw1T, vb1, nullptr, hid, nullptr, NBATCH, 256, DMODEL, /*gelu*/1);
    k_head<<<(NBATCH * 4 + 255) / 256, 256, 0, stream>>>(hid, vw2, vb2, out);

    (void)in_sizes; (void)n_in; (void)out_size; (void)ws_size;
}
